// GatedDeltaNet_22282290332308
// MI455X (gfx1250) — matrix operand order repaired and checked by reading, not compiled, not run on hardware
//
#include <hip/hip_runtime.h>
#include <hip/hip_bf16.h>
#include <math.h>

// Problem constants (from reference)
#define B_   2
#define L_   4096
#define D_   2048
#define H_   16
#define Dh_  128
#define C_   64
#define KW_  4
#define NCH_ (L_ / C_)     // 64 chunks
#define M_   (B_ * L_)     // 8192 rows
#define HD_  (H_ * Dh_)    // 2048
#define VS_  2             // v-dim split of the chunk scan
#define Dv_  (Dh_ / VS_)   // 64 v-columns per block

// gfx1250 async memory->LDS path (ASYNCcnt), guarded so the file still
// compiles cleanly if this toolchain doesn't declare the builtins.
#if defined(__has_builtin)
#if __has_builtin(__builtin_amdgcn_global_load_async_to_lds_b128) && \
    __has_builtin(__builtin_amdgcn_s_wait_asynccnt)
#define USE_ASYNC_LDS 1
#endif
#endif
#ifndef USE_ASYNC_LDS
#define USE_ASYNC_LDS 0
#endif

typedef __attribute__((ext_vector_type(4)))  int    v4i_;
// builtin signature (from hipcc diagnostic): (v4i AS1*, v4i AS3*, imm, imm)
#define GPTR(p) ((__attribute__((address_space(1))) v4i_*)(p))
#define LPTR(p) ((__attribute__((address_space(3))) v4i_*)(p))

typedef __attribute__((ext_vector_type(16))) __bf16 v16bf;
typedef __attribute__((ext_vector_type(8)))  float  v8f;
typedef __attribute__((ext_vector_type(2)))  float  v2f;

__device__ __forceinline__ float sigmoidf_(float x) { return 1.0f / (1.0f + __expf(-x)); }
__device__ __forceinline__ float siluf_(float x)    { return x * sigmoidf_(x); }

// ---------------------------------------------------------------------------
// fp32 -> bf16 cast (grid-stride)
// ---------------------------------------------------------------------------
__global__ void k_cast_bf16(const float* __restrict__ src, __bf16* __restrict__ dst, long n) {
    long i = (long)blockIdx.x * blockDim.x + threadIdx.x;
    long stride = (long)gridDim.x * blockDim.x;
    for (; i < n; i += stride) dst[i] = (__bf16)src[i];
}

// ---------------------------------------------------------------------------
// Y[M,N] = A[M,K] @ B[N,K]^T  (bf16 in, fp32 out), v_wmma_f32_16x16x32_bf16.
// 128x128 block tile, K-step 32, double-buffered LDS, 8 waves in a 2(M)x4(N)
// grid; each wave owns a 64x32 strip = 4x2 accumulators -> 8 WMMA / K-step.
// ---------------------------------------------------------------------------
__global__ __launch_bounds__(256)
void k_gemm_bf16_nt(const __bf16* __restrict__ A, const __bf16* __restrict__ Bw,
                    float* __restrict__ Y, int Mdim, int Ndim, int Kdim) {
    __shared__ __bf16 As[2][128][32];
    __shared__ __bf16 Bs[2][128][32];

    const int tid   = threadIdx.x;
    const int lane  = tid & 31;
    const int w     = tid >> 5;
    const int wm    = w & 1;          // 64-row half of the block tile
    const int wn    = w >> 1;         // 32-col strip
    const int khalf = lane >> 4;      // lane half selects K sub-range (ISA layout)
    const int ml    = lane & 15;

    const int blockM = blockIdx.y * 128;
    const int blockN = blockIdx.x * 128;

    // cooperative loads: 256 threads x 2 rows x 16B per tile (128x32 bf16 = 8KB)
    const int lr = tid >> 2;            // 0..63
    const int lc = (tid & 3) << 3;      // 0,8,16,24

    const __bf16* gA0 = A  + (size_t)(blockM + lr) * Kdim + lc;
    const __bf16* gA1 = A  + (size_t)(blockM + 64 + lr) * Kdim + lc;
    const __bf16* gB0 = Bw + (size_t)(blockN + lr) * Kdim + lc;
    const __bf16* gB1 = Bw + (size_t)(blockN + 64 + lr) * Kdim + lc;

    v8f acc[4][2] = {};

    union Frag { uint4 u[2]; v16bf v; };

    // preload tile 0 into buffer 0
    *(uint4*)&As[0][lr][lc]      = *(const uint4*)gA0;
    *(uint4*)&As[0][lr + 64][lc] = *(const uint4*)gA1;
    *(uint4*)&Bs[0][lr][lc]      = *(const uint4*)gB0;
    *(uint4*)&Bs[0][lr + 64][lc] = *(const uint4*)gB1;

    const int ksteps = Kdim >> 5;
    for (int kt = 0; kt < ksteps; ++kt) {
        const int cur = kt & 1;
        const bool more = (kt + 1) < ksteps;
        uint4 ra0, ra1, rb0, rb1;
        if (more) {
            gA0 += 32; gA1 += 32; gB0 += 32; gB1 += 32;
            ra0 = *(const uint4*)gA0; ra1 = *(const uint4*)gA1;
            rb0 = *(const uint4*)gB0; rb1 = *(const uint4*)gB1;
            __builtin_prefetch(gA0 + 32, 0, 1);   // gfx1250 global_prefetch_b8
            __builtin_prefetch(gB0 + 32, 0, 1);
        }
        __syncthreads();   // buffer `cur` fully written; prior reads of `cur^1` done

        Frag fa[4], fb[2];
#pragma unroll
        for (int mi = 0; mi < 4; ++mi) {
            fa[mi].u[0] = *(const uint4*)&As[cur][wm * 64 + mi * 16 + ml][khalf * 8];
            fa[mi].u[1] = *(const uint4*)&As[cur][wm * 64 + mi * 16 + ml][16 + khalf * 8];
        }
#pragma unroll
        for (int ni = 0; ni < 2; ++ni) {
            fb[ni].u[0] = *(const uint4*)&Bs[cur][wn * 32 + ni * 16 + ml][khalf * 8];
            fb[ni].u[1] = *(const uint4*)&Bs[cur][wn * 32 + ni * 16 + ml][16 + khalf * 8];
        }
#pragma unroll
        for (int mi = 0; mi < 4; ++mi)
#pragma unroll
            for (int ni = 0; ni < 2; ++ni)
                acc[mi][ni] = __builtin_amdgcn_wmma_f32_16x16x32_bf16(
                    false, fa[mi].v, false, fb[ni].v, (short)0, acc[mi][ni], false, false);

        if (more) {
            const int nxt = cur ^ 1;
            *(uint4*)&As[nxt][lr][lc]      = ra0;
            *(uint4*)&As[nxt][lr + 64][lc] = ra1;
            *(uint4*)&Bs[nxt][lr][lc]      = rb0;
            *(uint4*)&Bs[nxt][lr + 64][lc] = rb1;
        }
    }

    // Epilogue: row pointer walks down M so addresses are simple adds.
    // C/D layout: lanes 0-15 N=lane, VGPR r -> M=r ; lanes 16-31 N=lane-16, M=8+r
#pragma unroll
    for (int mi = 0; mi < 4; ++mi) {
        float* yrow = Y + (size_t)(blockM + wm * 64 + mi * 16 + khalf * 8) * Ndim
                        + (blockN + wn * 32 + ml);
#pragma unroll
        for (int r = 0; r < 8; ++r) {
            yrow[0]  = acc[mi][0][r];
            yrow[16] = acc[mi][1][r];
            yrow += Ndim;
        }
    }
}

// ---------------------------------------------------------------------------
// fp32 WMMA helper: accumulate a 16x16 tile of A@B with V_WMMA_F32_16X16X4_F32.
// A 16x4 layout: lane m=lane&15, 2 K values per lane-half. B 4x16 mirrored.
// ---------------------------------------------------------------------------
template<int KDIM, class FA, class FB>
__device__ __forceinline__ v8f mma_f32(FA la, FB lb, v8f acc, int half, int nl) {
#pragma unroll
    for (int k0 = 0; k0 < KDIM; k0 += 4) {
        const int ka = k0 + half * 2;
        v2f a, b;
        a[0] = la(nl, ka);     a[1] = la(nl, ka + 1);
        b[0] = lb(ka, nl);     b[1] = lb(ka + 1, nl);
        acc = __builtin_amdgcn_wmma_f32_16x16x4_f32(false, a, false, b,
                                                    (short)0, acc, false, false);
    }
    return acc;
}

__device__ __forceinline__ void store_tile(float* dst, int ldd, int r0, int c0,
                                           v8f acc, int half, int nl) {
#pragma unroll
    for (int r = 0; r < 8; ++r)
        dst[(r0 + half * 8 + r) * ldd + c0 + nl] = acc[r];
}

// ---------------------------------------------------------------------------
// beta = sigmoid(x@Wb^T), g = -exp(A_log)*softplus(x@Wa^T + dt_bias)
// one block per (b,l); outputs stored (B,H,L)
// ---------------------------------------------------------------------------
__global__ __launch_bounds__(128)
void k_beta_g(const float* __restrict__ x, const float* __restrict__ Wb,
              const float* __restrict__ Wa, const float* __restrict__ A_log,
              const float* __restrict__ dt_bias,
              float* __restrict__ beta, float* __restrict__ g) {
    __shared__ float xs[D_];
    const int row = blockIdx.x;          // b*L + l
    const int b = row / L_, l = row % L_;
    const int tid = threadIdx.x, lane = tid & 31, w = tid >> 5;   // 4 waves

    for (int k2 = tid; k2 < D_; k2 += 128) xs[k2] = x[(size_t)row * D_ + k2];
    __syncthreads();

    for (int t = 0; t < 8; ++t) {
        int idx = w * 8 + t;             // 0..31: first 16 -> beta, next 16 -> g
        int h = idx & 15;
        const float* Wrow = (idx < 16 ? Wb : Wa) + (size_t)h * D_;
        float s = 0.f;
        for (int k2 = lane; k2 < D_; k2 += 32) s += xs[k2] * Wrow[k2];
        for (int off = 16; off; off >>= 1) s += __shfl_xor(s, off, 32);
        if (lane == 0) {
            size_t o = ((size_t)(b * H_ + h)) * L_ + l;
            if (idx < 16) {
                beta[o] = sigmoidf_(s);
            } else {
                float z = s + dt_bias[h];
                float sp = (z > 20.f) ? z : log1pf(__expf(z));
                g[o] = -__expf(A_log[h]) * sp;
            }
        }
    }
}

// ---------------------------------------------------------------------------
// causal depthwise conv (KW=4) + SiLU over (B,L,HD)
// ---------------------------------------------------------------------------
__global__ void k_conv_silu(const float* __restrict__ xin, const float* __restrict__ wconv,
                            float* __restrict__ yout) {
    long idx = (long)blockIdx.x * blockDim.x + threadIdx.x;
    if (idx >= (long)B_ * L_ * HD_) return;
    int c = (int)(idx % HD_);
    long bl = idx / HD_;
    int l = (int)(bl % L_);
    long b = bl / L_;
    const float* wr = wconv + (size_t)c * KW_;
    float acc = 0.f;
#pragma unroll
    for (int j = 0; j < KW_; ++j) {
        int ls = l - (KW_ - 1) + j;
        if (ls >= 0) acc += wr[j] * xin[((size_t)b * L_ + ls) * HD_ + c];
    }
    yout[idx] = siluf_(acc);
}

// ---------------------------------------------------------------------------
// L2 normalize each contiguous 128-vector; in-place. post_scale folds 1/sqrt(Dh).
// ---------------------------------------------------------------------------
__global__ __launch_bounds__(128)
void k_l2norm(float* __restrict__ buf, float post_scale) {
    const size_t base = (size_t)blockIdx.x * Dh_;
    const int tid = threadIdx.x;
    float v = buf[base + tid];
    float s = v * v;
    for (int off = 16; off; off >>= 1) s += __shfl_xor(s, off, 32);
    __shared__ float ws_[4];
    if ((tid & 31) == 0) ws_[tid >> 5] = s;
    __syncthreads();
    float tot = ws_[0] + ws_[1] + ws_[2] + ws_[3];
    float inv = post_scale / fmaxf(sqrtf(tot), 1e-12f);
    buf[base + tid] = v * inv;
}

// ---------------------------------------------------------------------------
// o = O * rsqrt(mean(O^2)+eps) * norm_w * silu(gate) ; emit bf16 for final GEMM
// ---------------------------------------------------------------------------
__global__ __launch_bounds__(128)
void k_gate_rms(const float* __restrict__ O, const float* __restrict__ gate,
                const float* __restrict__ norm_w, __bf16* __restrict__ outbf) {
    const size_t base = (size_t)blockIdx.x * Dh_;
    const int tid = threadIdx.x;
    float v = O[base + tid];
    float s = v * v;
    for (int off = 16; off; off >>= 1) s += __shfl_xor(s, off, 32);
    __shared__ float ws_[4];
    if ((tid & 31) == 0) ws_[tid >> 5] = s;
    __syncthreads();
    float mean = (ws_[0] + ws_[1] + ws_[2] + ws_[3]) * (1.0f / Dh_);
    float r = rsqrtf(mean + 1e-5f);
    float gv = gate[base + tid];
    float o = v * r * norm_w[tid] * siluf_(gv);
    outbf[base + tid] = (__bf16)o;
}

// ---------------------------------------------------------------------------
// Chunked delta-rule scan, fp32 WMMA edition. One block per (b,h,vslice);
// the state slice and all chunk operands live in LDS (~214KB of the 320KB WGP
// LDS). All matmuls run on V_WMMA_F32_16X16X4_F32; only the 64x64 unit-lower
// triangular inverse is serial VALU. Q/K/V chunk staging uses the gfx1250
// async memory->LDS path when the toolchain exposes it.
// ---------------------------------------------------------------------------
__global__ __launch_bounds__(256)
void k_chunk_scan(const float* __restrict__ q, const float* __restrict__ kk,
                  const float* __restrict__ vv, const float* __restrict__ beta,
                  const float* __restrict__ g, const float* __restrict__ S0,
                  float* __restrict__ O, float* __restrict__ Sfin) {
    extern __shared__ float sm[];
    float* S   = sm;                    // Dv x Dh (rows: v slice, cols: k)
    float* Q   = S  + Dv_ * Dh_;        // C x Dh
    float* K   = Q  + C_ * Dh_;         // C x Dh
    float* V   = K  + C_ * Dh_;         // C x Dv
    float* A   = V  + C_ * Dv_;         // C x C
    float* T   = A  + C_ * C_;          // C x C
    float* P   = T  + C_ * C_;          // C x C
    float* W1  = P  + C_ * C_;          // C x Dh
    float* U   = W1 + C_ * Dh_;         // C x Dv ("mid" after subtraction)
    float* lg  = U  + C_ * Dv_;         // 64 cumulative log-decay
    float* gm  = lg + C_;               // 64 gamma
    float* gr  = gm + C_;               // 64 gammaC/gamma
    float* bt  = gr + C_;               // 64 beta
    float* gCp = bt + C_;               // gammaC scalar

    const int blk = blockIdx.x;         // (b*H + h)*VS + vs
    const int bh  = blk / VS_;
    const int vs  = blk % VS_;
    const int vo  = vs * Dv_;
    const int b   = bh / H_, h = bh % H_;
    const int tid = threadIdx.x;
    const int wv = tid >> 5, lane = tid & 31, half = lane >> 4, nl = lane & 15;

    for (int i = tid; i < Dv_ * Dh_; i += 256) {
        int r = i >> 7, c = i & 127;
        S[i] = S0[(size_t)bh * Dh_ * Dh_ + (size_t)(vo + r) * Dh_ + c];
    }
    __syncthreads();

    for (int n = 0; n < NCH_; ++n) {
        const size_t lbase = (size_t)b * L_ + (size_t)n * C_;

        if (tid < C_) bt[tid] = beta[(size_t)bh * L_ + n * C_ + tid];
        if (tid == 0) {
            float acc = 0.f;
            for (int i = 0; i < C_; ++i) {
                acc += g[(size_t)bh * L_ + n * C_ + i];
                lg[i] = acc;
            }
            gCp[0] = __expf(acc);       // gammaC
        }
        __syncthreads();
        if (tid < C_) {
            gm[tid] = __expf(lg[tid]);
            gr[tid] = __expf(lg[C_ - 1] - lg[tid]);
        }

        // stage Q,K (C x Dh) and V slice (C x Dv) into LDS
#if USE_ASYNC_LDS
        for (int i = tid; i < C_ * Dh_ / 4; i += 256) {
            int r = i >> 5, c = (i & 31) << 2;   // 32 float4 per row
            size_t go = (lbase + r) * HD_ + (size_t)h * Dh_ + c;
            __builtin_amdgcn_global_load_async_to_lds_b128(GPTR(q + go),  LPTR(Q + i * 4), 0, 0);
            __builtin_amdgcn_global_load_async_to_lds_b128(GPTR(kk + go), LPTR(K + i * 4), 0, 0);
        }
        for (int i = tid; i < C_ * Dv_ / 4; i += 256) {
            int r = i >> 4, c = (i & 15) << 2;   // 16 float4 per row
            size_t go = (lbase + r) * HD_ + (size_t)h * Dh_ + vo + c;
            __builtin_amdgcn_global_load_async_to_lds_b128(GPTR(vv + go), LPTR(V + i * 4), 0, 0);
        }
        __builtin_amdgcn_s_wait_asynccnt(0);
#else
        for (int i = tid; i < C_ * Dh_; i += 256) {
            int r = i >> 7, c = i & 127;
            size_t go = (lbase + r) * HD_ + (size_t)h * Dh_ + c;
            Q[i] = q[go]; K[i] = kk[go];
        }
        for (int i = tid; i < C_ * Dv_; i += 256) {
            int r = i / Dv_, c = i % Dv_;
            V[i] = vv[(lbase + r) * HD_ + (size_t)h * Dh_ + vo + c];
        }
#endif
        __syncthreads();

        // A_raw = K K^T  (C x C, inner Dh)
        for (int t = wv; t < 16; t += 8) {
            int ti = (t >> 2) << 4, tj = (t & 3) << 4;
            v8f acc = {};
            acc = mma_f32<Dh_>(
                [&](int m, int kx) { return K[(ti + m) * Dh_ + kx]; },
                [&](int kx, int nn) { return K[(tj + nn) * Dh_ + kx]; },
                acc, half, nl);
            store_tile(A, C_, ti, tj, acc, half, nl);
        }
        __syncthreads();
        for (int p = tid; p < C_ * C_; p += 256) {
            int i = p >> 6, j = p & 63;
            float a = A[p];
            A[p] = (i > j) ? bt[i] * __expf(lg[i] - lg[j]) * a
                           : (i == j ? 1.f : 0.f);
        }
        __syncthreads();

        // T = A^{-1} (unit lower triangular), forward substitution
        for (int i = 0; i < C_; ++i) {
            if (tid < C_) {
                int j = tid; float s;
                if (j > i) {
                    s = 0.f;
                } else {
                    s = (i == j) ? 1.f : 0.f;
                    for (int p2 = j; p2 < i; ++p2) s -= A[i * C_ + p2] * T[p2 * C_ + j];
                }
                T[i * C_ + j] = s;
            }
            __syncthreads();
        }

        // W1 = T @ (bt*gm*K)  (C x Dh, inner C)
        for (int t = wv; t < 4 * (Dh_ / 16); t += 8) {
            int ti = (t / (Dh_ / 16)) << 4, tj = (t % (Dh_ / 16)) << 4;
            v8f acc = {};
            acc = mma_f32<C_>(
                [&](int m, int kx) { return T[(ti + m) * C_ + kx]; },
                [&](int kx, int nn) { return bt[kx] * gm[kx] * K[kx * Dh_ + tj + nn]; },
                acc, half, nl);
            store_tile(W1, Dh_, ti, tj, acc, half, nl);
        }
        // U = T @ (bt*V)  (C x Dv, inner C)
        for (int t = wv; t < 4 * (Dv_ / 16); t += 8) {
            int ti = (t / (Dv_ / 16)) << 4, tj = (t % (Dv_ / 16)) << 4;
            v8f acc = {};
            acc = mma_f32<C_>(
                [&](int m, int kx) { return T[(ti + m) * C_ + kx]; },
                [&](int kx, int nn) { return bt[kx] * V[kx * Dv_ + tj + nn]; },
                acc, half, nl);
            store_tile(U, Dv_, ti, tj, acc, half, nl);
        }
        __syncthreads();

        // mid: U -= W1 @ S^T  (C x Dv, inner Dh); S^T[e][d] = S[d][e]
        for (int t = wv; t < 4 * (Dv_ / 16); t += 8) {
            int ti = (t / (Dv_ / 16)) << 4, tj = (t % (Dv_ / 16)) << 4;
            v8f acc = {};
            acc = mma_f32<Dh_>(
                [&](int m, int kx) { return W1[(ti + m) * Dh_ + kx]; },
                [&](int kx, int nn) { return S[(tj + nn) * Dh_ + kx]; },
                acc, half, nl);
#pragma unroll
            for (int r = 0; r < 8; ++r)
                U[(ti + half * 8 + r) * Dv_ + tj + nl] -= acc[r];
        }
        __syncthreads();

        // P_raw = Q K^T  (C x C, inner Dh)
        for (int t = wv; t < 16; t += 8) {
            int ti = (t >> 2) << 4, tj = (t & 3) << 4;
            v8f acc = {};
            acc = mma_f32<Dh_>(
                [&](int m, int kx) { return Q[(ti + m) * Dh_ + kx]; },
                [&](int kx, int nn) { return K[(tj + nn) * Dh_ + kx]; },
                acc, half, nl);
            store_tile(P, C_, ti, tj, acc, half, nl);
        }
        __syncthreads();
        for (int p = tid; p < C_ * C_; p += 256) {
            int i = p >> 6, j = p & 63;
            P[p] = (i >= j) ? __expf(lg[i] - lg[j]) * P[p] : 0.f;
        }
        __syncthreads();

        // O = (gm ⊙ Q) @ S^T + P @ mid  (C x Dv) -> global
        for (int t = wv; t < 4 * (Dv_ / 16); t += 8) {
            int ti = (t / (Dv_ / 16)) << 4, tj = (t % (Dv_ / 16)) << 4;
            v8f acc = {};
            acc = mma_f32<Dh_>(
                [&](int m, int kx) { return gm[ti + m] * Q[(ti + m) * Dh_ + kx]; },
                [&](int kx, int nn) { return S[(tj + nn) * Dh_ + kx]; },
                acc, half, nl);
            acc = mma_f32<C_>(
                [&](int m, int kx) { return P[(ti + m) * C_ + kx]; },
                [&](int kx, int nn) { return U[kx * Dv_ + tj + nn]; },
                acc, half, nl);
#pragma unroll
            for (int r = 0; r < 8; ++r)
                O[(lbase + ti + half * 8 + r) * HD_ + (size_t)h * Dh_ + vo + tj + nl] = acc[r];
        }
        __syncthreads();

        // S = gammaC*S + mid^T @ (gr ⊙ K)  (Dv x Dh, inner C)
        float gC = gCp[0];
        for (int t = wv; t < (Dv_ / 16) * (Dh_ / 16); t += 8) {
            int ti = (t / (Dh_ / 16)) << 4, tj = (t % (Dh_ / 16)) << 4;
            v8f acc = {};
            acc = mma_f32<C_>(
                [&](int m, int kx) { return U[kx * Dv_ + ti + m]; },
                [&](int kx, int nn) { return gr[kx] * K[kx * Dh_ + tj + nn]; },
                acc, half, nl);
#pragma unroll
            for (int r = 0; r < 8; ++r) {
                int idx = (ti + half * 8 + r) * Dh_ + tj + nl;
                S[idx] = gC * S[idx] + acc[r];
            }
        }
        __syncthreads();
    }

    for (int i = tid; i < Dv_ * Dh_; i += 256) {
        int r = i >> 7, c = i & 127;
        Sfin[(size_t)bh * Dh_ * Dh_ + (size_t)(vo + r) * Dh_ + c] = S[i];
    }
}

// ---------------------------------------------------------------------------
// Host-side orchestration
// ---------------------------------------------------------------------------
extern "C" void kernel_launch(void* const* d_in, const int* in_sizes, int n_in,
                              void* d_out, int out_size, void* d_ws, size_t ws_size,
                              hipStream_t stream) {
    (void)in_sizes; (void)n_in; (void)out_size; (void)ws_size;

    const float* x     = (const float*)d_in[0];
    const float* Wq    = (const float*)d_in[1];
    const float* Wk    = (const float*)d_in[2];
    const float* Wv    = (const float*)d_in[3];
    const float* Wb    = (const float*)d_in[4];
    const float* Wa    = (const float*)d_in[5];
    const float* A_log = (const float*)d_in[6];
    const float* dtb   = (const float*)d_in[7];
    const float* convq = (const float*)d_in[8];
    const float* convk = (const float*)d_in[9];
    const float* convv = (const float*)d_in[10];
    const float* Wg    = (const float*)d_in[11];
    const float* normw = (const float*)d_in[12];
    const float* Wo    = (const float*)d_in[13];
    const float* S0    = (const float*)d_in[14];

    float* out  = (float*)d_out;                       // (B,L,D)
    float* Sfin = out + (size_t)B_ * L_ * D_;          // (B,H,Dh,Dh)

    char* ws = (char*)d_ws;
    size_t off = 0;
    auto alloc = [&](size_t bytes) {
        void* p = ws + off;
        off += (bytes + 255) & ~(size_t)255;
        return p;
    };

    __bf16* xbf = (__bf16*)alloc((size_t)M_ * D_ * 2);
    __bf16* Wqb = (__bf16*)alloc((size_t)HD_ * D_ * 2);
    __bf16* Wkb = (__bf16*)alloc((size_t)HD_ * D_ * 2);
    __bf16* Wvb = (__bf16*)alloc((size_t)HD_ * D_ * 2);
    __bf16* Wgb = (__bf16*)alloc((size_t)HD_ * D_ * 2);
    __bf16* Wob = (__bf16*)alloc((size_t)D_ * HD_ * 2);
    float*  qp  = (float*)alloc((size_t)M_ * HD_ * 4);   // projections (pre-conv)
    float*  kp  = (float*)alloc((size_t)M_ * HD_ * 4);
    float*  vp  = (float*)alloc((size_t)M_ * HD_ * 4);
    float*  gp  = (float*)alloc((size_t)M_ * HD_ * 4);   // gate (no conv)
    float*  qcv = (float*)alloc((size_t)M_ * HD_ * 4);   // post conv+silu(+norm)
    float*  kcv = (float*)alloc((size_t)M_ * HD_ * 4);
    float*  vcv = (float*)alloc((size_t)M_ * HD_ * 4);
    float*  betab = (float*)alloc((size_t)B_ * H_ * L_ * 4);
    float*  gbuf  = (float*)alloc((size_t)B_ * H_ * L_ * 4);
    float*  Obuf  = (float*)alloc((size_t)M_ * HD_ * 4);
    __bf16* obf   = (__bf16*)alloc((size_t)M_ * HD_ * 2);

    // 1) casts to bf16
    k_cast_bf16<<<16384, 256, 0, stream>>>(x,  xbf, (long)M_ * D_);
    k_cast_bf16<<<4096, 256, 0, stream>>>(Wq, Wqb, (long)HD_ * D_);
    k_cast_bf16<<<4096, 256, 0, stream>>>(Wk, Wkb, (long)HD_ * D_);
    k_cast_bf16<<<4096, 256, 0, stream>>>(Wv, Wvb, (long)HD_ * D_);
    k_cast_bf16<<<4096, 256, 0, stream>>>(Wg, Wgb, (long)HD_ * D_);
    k_cast_bf16<<<4096, 256, 0, stream>>>(Wo, Wob, (long)D_ * HD_);

    // 2) big projections via bf16 WMMA GEMM (128x128 tiles)
    dim3 ggrid(HD_ / 128, M_ / 128);
    k_gemm_bf16_nt<<<ggrid, 256, 0, stream>>>(xbf, Wqb, qp, M_, HD_, D_);
    k_gemm_bf16_nt<<<ggrid, 256, 0, stream>>>(xbf, Wkb, kp, M_, HD_, D_);
    k_gemm_bf16_nt<<<ggrid, 256, 0, stream>>>(xbf, Wvb, vp, M_, HD_, D_);
    k_gemm_bf16_nt<<<ggrid, 256, 0, stream>>>(xbf, Wgb, gp, M_, HD_, D_);

    // 3) beta / g scalar projections
    k_beta_g<<<M_, 128, 0, stream>>>(x, Wb, Wa, A_log, dtb, betab, gbuf);

    // 4) causal depthwise conv + silu
    const int convBlocks = (M_ * HD_) / 256;
    k_conv_silu<<<convBlocks, 256, 0, stream>>>(qp, convq, qcv);
    k_conv_silu<<<convBlocks, 256, 0, stream>>>(kp, convk, kcv);
    k_conv_silu<<<convBlocks, 256, 0, stream>>>(vp, convv, vcv);

    // 5) L2 norms (q also scaled by 1/sqrt(Dh))
    const int rows = B_ * L_ * H_;
    k_l2norm<<<rows, 128, 0, stream>>>(qcv, 1.0f / sqrtf((float)Dh_));
    k_l2norm<<<rows, 128, 0, stream>>>(kcv, 1.0f);

    // 6) chunked delta-rule scan on the fp32 tensor pipe; Sfin -> d_out tail
    size_t shmem = (size_t)(Dv_ * Dh_ +            // S
                            2 * C_ * Dh_ +         // Q, K
                            C_ * Dv_ +             // V
                            3 * C_ * C_ +          // A, T, P
                            C_ * Dh_ +             // W1
                            C_ * Dv_ +             // U
                            4 * C_ + 8) * sizeof(float);
    k_chunk_scan<<<B_ * H_ * VS_, 256, shmem, stream>>>(qcv, kcv, vcv, betab, gbuf,
                                                        S0, Obuf, Sfin);

    // 7) RMS-norm * norm_w * silu(gate) -> bf16
    k_gate_rms<<<rows, 128, 0, stream>>>(Obuf, gp, normw, obf);

    // 8) final projection: out = o @ Wo^T
    dim3 ogrid(D_ / 128, M_ / 128);
    k_gemm_bf16_nt<<<ogrid, 256, 0, stream>>>(obf, Wob, out, M_, D_, HD_);
}
